// ccn_46127948759816
// MI455X (gfx1250) — matrix operand order repaired and checked by reading, not compiled, not run on hardware
//
#include <hip/hip_runtime.h>
#include <hip/hip_bf16.h>
#include <math.h>

typedef __attribute__((ext_vector_type(16))) _Float16 v16h;
typedef __attribute__((ext_vector_type(8)))  float    v8f;

#define B_    8
#define N_    2048
#define KNN_  20
#define EPS_  1e-5f
#define NEG_INF_ (-3.4e38f)

// ---------------- WMMA fragment index helpers (CDNA5 16x16x32 f16, wave32) ---
// A (16x32 f16): lane<16 row M=lane, K-halves {0..7,16..23}; lane>=16 same row,
// K-halves {8..15,24..31}. v16h element e -> VGPR v=e>>1, pos=e&1.
__device__ __forceinline__ int wmma_a_k(int lane, int e) {
  int v = e >> 1, p = e & 1;
  int base = (lane & 16) ? 8 : 0;
  int off  = (v < 4) ? (2 * v) : (2 * v + 8);
  return base + off + p;
}
// B (32x16 f16): lane%16 = column N; lanes<16 K={0..7,16..23}, lanes>=16
// K={8..15,24..31} (same K order as A).
__device__ __forceinline__ int wmma_b_k(int lane, int e) {
  int v = e >> 1, p = e & 1;
  int base = (lane & 16) ? 8 : 0;
  int off  = (v < 4) ? (2 * v) : (2 * v + 8);
  return base + off + p;
}
// C/D (f32): col = lane&15, row = r + ((lane&16)?8:0)

__device__ __forceinline__ float lrelu(float x) { return x > 0.f ? x : 0.2f * x; }

// ---------------- kNN (top-K by -||xi-xj||^2, jax top_k tie-break) ----------
template <int C, int TILE>
__global__ void knn_kernel(const float* __restrict__ src, int* __restrict__ idx) {
  __shared__ float tile[C][TILE];
  __shared__ float xxs[TILE];
  int b = blockIdx.y;
  int n = blockIdx.x * blockDim.x + threadIdx.x;
  const float* sb = src + (size_t)b * C * N_;
  float ctr[C];
#pragma unroll
  for (int c = 0; c < C; c++) ctr[c] = sb[c * N_ + n];
  float bv[KNN_]; int bi[KNN_];
#pragma unroll
  for (int k = 0; k < KNN_; k++) { bv[k] = NEG_INF_; bi[k] = 0; }
  for (int j0 = 0; j0 < N_; j0 += TILE) {
    __syncthreads();
    for (int t = threadIdx.x; t < C * TILE; t += blockDim.x) {
      int c = t / TILE, j = t % TILE;
      tile[c][j] = sb[c * N_ + j0 + j];
    }
    __syncthreads();
    for (int t = threadIdx.x; t < TILE; t += blockDim.x) {
      float s = 0.f;
#pragma unroll
      for (int c = 0; c < C; c++) { float x = tile[c][t]; s += x * x; }
      xxs[t] = s;
    }
    __syncthreads();
    for (int j = 0; j < TILE; j++) {
      float d = 0.f;
#pragma unroll
      for (int c = 0; c < C; c++) d += ctr[c] * tile[c][j];
      float s = 2.f * d - xxs[j];  // rank-equivalent to pd (monotone shift)
      if (s > bv[KNN_ - 1]) {
        int p = KNN_ - 1;
        while (p > 0 && s > bv[p - 1]) { bv[p] = bv[p - 1]; bi[p] = bi[p - 1]; p--; }
        bv[p] = s; bi[p] = j0 + j;
      }
    }
  }
  int* o = idx + ((size_t)(b * N_ + n)) * KNN_;
#pragma unroll
  for (int k = 0; k < KNN_; k++) o[k] = bi[k];
}

// ---------------- transform net: conv1(64x6)+conv2(128x64) + max_k ----------
__global__ void tnet_edge_kernel(const float* __restrict__ x, const int* __restrict__ idx,
    const float* __restrict__ W1, const float* g1, const float* b1, const float* m1, const float* v1,
    const float* __restrict__ W2, const float* g2, const float* b2, const float* m2, const float* v2,
    float* __restrict__ outmax /*(B,128,N), written fully*/) {
  __shared__ _Float16 h1[320][64];
  __shared__ float sbuf[320 * 16];  // phase A: edge6[320][6]; phase B: obuf[320][16]
  int bb = blockIdx.x / (N_ / 16);
  int n0 = (blockIdx.x % (N_ / 16)) * 16;
  int o0 = blockIdx.y * 16;
  const float* xb = x + (size_t)bb * 3 * N_;
  const int* ib = idx + ((size_t)bb * N_ + n0) * KNN_;
  for (int t = threadIdx.x; t < 320; t += blockDim.x) {
    int nl = t / KNN_, k = t % KNN_;
    int j = ib[nl * KNN_ + k];
    int n = n0 + nl;
#pragma unroll
    for (int c = 0; c < 3; c++) {
      float cv = xb[c * N_ + n];
      sbuf[t * 6 + c]     = xb[c * N_ + j] - cv;
      sbuf[t * 6 + 3 + c] = cv;
    }
  }
  __syncthreads();
  for (int t = threadIdx.x; t < 320 * 64; t += blockDim.x) {
    int r = t >> 6, oc = t & 63;
    float acc = 0.f;
#pragma unroll
    for (int c = 0; c < 6; c++) acc += W1[oc * 6 + c] * sbuf[r * 6 + c];
    float s = g1[oc] * rsqrtf(v1[oc] + EPS_);
    float z = acc * s + (b1[oc] - m1[oc] * s);
    h1[r][oc] = (_Float16)lrelu(z);
  }
  __syncthreads();  // also fences sbuf reuse as obuf
  int wave = threadIdx.x >> 5, lane = threadIdx.x & 31;
  int col = lane & 15;
  int o = o0 + col;
  float s2 = g2[o] * rsqrtf(v2[o] + EPS_);
  float sh2 = b2[o] - m2[o] * s2;
  v16h bfr[2];
#pragma unroll
  for (int kc = 0; kc < 2; kc++)
#pragma unroll
    for (int e = 0; e < 16; e++)
      bfr[kc][e] = (_Float16)W2[o * 64 + kc * 32 + wmma_b_k(lane, e)];
  int rbase = (lane & 16) ? 8 : 0;
  for (int tile = wave; tile < 20; tile += 4) {
    int r = tile * 16 + col;
    v8f acc = {};
#pragma unroll
    for (int kc = 0; kc < 2; kc++) {
      v16h a;
#pragma unroll
      for (int e = 0; e < 16; e++) a[e] = h1[r][kc * 32 + wmma_a_k(lane, e)];
      acc = __builtin_amdgcn_wmma_f32_16x16x32_f16(false, a, false, bfr[kc], (short)0, acc, false, false);
    }
#pragma unroll
    for (int rr = 0; rr < 8; rr++) {
      int row = tile * 16 + rr + rbase;
      sbuf[row * 16 + col] = lrelu(acc[rr] * s2 + sh2);
    }
  }
  __syncthreads();
  for (int t = threadIdx.x; t < 256; t += blockDim.x) {
    int nl = t >> 4, oc = t & 15;
    float mx = NEG_INF_;
#pragma unroll
    for (int k = 0; k < KNN_; k++) mx = fmaxf(mx, sbuf[(nl * KNN_ + k) * 16 + oc]);
    outmax[((size_t)bb * 128 + o0 + oc) * N_ + n0 + nl] = mx;
  }
}

// ---------------- transform net: conv3 (1024x128), partial max over N -------
__global__ void tnet_conv3_kernel(const float* __restrict__ in /*(B,128,N)*/,
    const float* __restrict__ W /*1024x128*/,
    const float* g, const float* bb_, const float* m, const float* v,
    float* __restrict__ pmax /*(B,1024,N/16,2)*/) {
  int b = blockIdx.x / (N_ / 16);
  int nch = blockIdx.x % (N_ / 16);
  int n0 = nch * 16;
  int o0 = blockIdx.y * 16;
  int lane = threadIdx.x & 31;
  int col = lane & 15;
  int o = o0 + col;
  const float* ib = in + (size_t)b * 128 * N_;
  v8f acc = {};
#pragma unroll
  for (int kc = 0; kc < 4; kc++) {
    v16h a, bf;
#pragma unroll
    for (int e = 0; e < 16; e++) {
      int ka = kc * 32 + wmma_a_k(lane, e);
      a[e]  = (_Float16)ib[(size_t)ka * N_ + n0 + col];
      bf[e] = (_Float16)W[o * 128 + kc * 32 + wmma_b_k(lane, e)];
    }
    acc = __builtin_amdgcn_wmma_f32_16x16x32_f16(false, a, false, bf, (short)0, acc, false, false);
  }
  float s = g[o] * rsqrtf(v[o] + EPS_);
  float sh = bb_[o] - m[o] * s;
  float mx = NEG_INF_;
#pragma unroll
  for (int r = 0; r < 8; r++) mx = fmaxf(mx, lrelu(acc[r] * s + sh));
  int half = (lane & 16) ? 1 : 0;
  pmax[(((size_t)b * 1024 + o) * (N_ / 16) + nch) * 2 + half] = mx;
}

// ---------------- fixed-order partial reductions (deterministic) ------------
__global__ void reduce_max_kernel(const float* __restrict__ pin, float* __restrict__ out) {
  int i = blockIdx.x * blockDim.x + threadIdx.x;  // over B*1024
  if (i < B_ * 1024) {
    const float* p = pin + (size_t)i * (N_ / 16) * 2;
    float mx = NEG_INF_;
    for (int c = 0; c < (N_ / 16) * 2; c++) mx = fmaxf(mx, p[c]);
    out[i] = mx;
  }
}

__global__ void reduce_maxsum_kernel(const float* __restrict__ pmax, const float* __restrict__ psum,
                                     float* __restrict__ gmax, float* __restrict__ gsum) {
  int i = blockIdx.x * blockDim.x + threadIdx.x;  // over B*1024
  if (i < B_ * 1024) {
    const float* pm = pmax + (size_t)i * (N_ / 16) * 2;
    const float* ps = psum + (size_t)i * (N_ / 16) * 2;
    float mx = NEG_INF_, sm = 0.f;
    for (int c = 0; c < (N_ / 16) * 2; c++) { mx = fmaxf(mx, pm[c]); sm += ps[c]; }
    gmax[i] = mx;
    gsum[i] = sm;
  }
}

// ---------------- transform net MLP head ------------------------------------
__global__ void tnet_mlp_kernel(const float* __restrict__ tg,
    const float* __restrict__ L1, const float* g3, const float* b3, const float* m3, const float* v3,
    const float* __restrict__ L2, const float* g4, const float* b4, const float* m4, const float* v4,
    const float* __restrict__ TW, const float* __restrict__ TB, float* __restrict__ tmat) {
  __shared__ float h0[1024], h1s[512], h2s[256];
  int b = blockIdx.x;
  for (int t = threadIdx.x; t < 1024; t += blockDim.x) h0[t] = tg[b * 1024 + t];
  __syncthreads();
  for (int o = threadIdx.x; o < 512; o += blockDim.x) {
    float acc = 0.f;
    for (int c = 0; c < 1024; c++) acc += L1[o * 1024 + c] * h0[c];
    float s = g3[o] * rsqrtf(v3[o] + EPS_);
    h1s[o] = lrelu(acc * s + (b3[o] - m3[o] * s));
  }
  __syncthreads();
  for (int o = threadIdx.x; o < 256; o += blockDim.x) {
    float acc = 0.f;
    for (int c = 0; c < 512; c++) acc += L2[o * 512 + c] * h1s[c];
    float s = g4[o] * rsqrtf(v4[o] + EPS_);
    h2s[o] = lrelu(acc * s + (b4[o] - m4[o] * s));
  }
  __syncthreads();
  if (threadIdx.x < 9) {
    float acc = TB[threadIdx.x];
    for (int c = 0; c < 256; c++) acc += TW[threadIdx.x * 256 + c] * h2s[c];
    tmat[b * 9 + threadIdx.x] = acc;
  }
}

// ---------------- apply 3x3 transform: out[d][n] = sum_c x[c][n]*t[c][d] ----
__global__ void rotate_kernel(const float* __restrict__ x, const float* __restrict__ tmat,
                              float* __restrict__ xr) {
  int b = blockIdx.y;
  int n = blockIdx.x * blockDim.x + threadIdx.x;
  const float* xb = x + (size_t)b * 3 * N_;
  const float* t = tmat + b * 9;
  float p0 = xb[n], p1 = xb[N_ + n], p2 = xb[2 * N_ + n];
#pragma unroll
  for (int d = 0; d < 3; d++)
    xr[((size_t)b * 3 + d) * N_ + n] = p0 * t[d] + p1 * t[3 + d] + p2 * t[6 + d];
}

// ---------------- edge conv: cbl(edge(src,idx), W, bn).max_k ----------------
template <int CIN, int COUT>
__global__ void edgeconv_kernel(const float* __restrict__ src, const int* __restrict__ idx,
    const float* __restrict__ W /*COUT x 2CIN*/,
    const float* g, const float* bb_, const float* m, const float* v,
    float* __restrict__ outmax /*(B,COUT,N), written fully*/) {
  constexpr int CE = 2 * CIN;
  constexpr int KCH = (CE + 31) / 32;
  __shared__ _Float16 nbf[320][CIN];  // feat - center
  __shared__ _Float16 ctrs[16][CIN];
  __shared__ float obuf[320][16];
  int b = blockIdx.x / (N_ / 16);
  int n0 = (blockIdx.x % (N_ / 16)) * 16;
  int o0 = blockIdx.y * 16;
  const float* sb = src + (size_t)b * CIN * N_;
  const int* ib = idx + ((size_t)b * N_ + n0) * KNN_;
  // warm L2/WGP$ for the weight tile reused by all blocks
  for (int t = threadIdx.x * 64; t < COUT * CE * 4; t += blockDim.x * 64)
    __builtin_prefetch((const char*)W + t, 0, 0);
  for (int t = threadIdx.x; t < 16 * CIN; t += blockDim.x) {
    int nl = t / CIN, c = t % CIN;
    ctrs[nl][c] = (_Float16)sb[c * N_ + n0 + nl];
  }
  for (int t = threadIdx.x; t < 320 * CIN; t += blockDim.x) {
    int r = t / CIN, c = t % CIN;
    int nl = r / KNN_, k = r % KNN_;
    int j = ib[nl * KNN_ + k];
    nbf[r][c] = (_Float16)(sb[c * N_ + j] - sb[c * N_ + n0 + nl]);
  }
  __syncthreads();
  int wave = threadIdx.x >> 5, lane = threadIdx.x & 31;
  int col = lane & 15;
  int o = o0 + col;
  float s = g[o] * rsqrtf(v[o] + EPS_);
  float sh = bb_[o] - m[o] * s;
  v16h bfr[KCH];
#pragma unroll
  for (int kc = 0; kc < KCH; kc++)
#pragma unroll
    for (int e = 0; e < 16; e++) {
      int kb = kc * 32 + wmma_b_k(lane, e);
      bfr[kc][e] = (kb < CE) ? (_Float16)W[o * CE + kb] : (_Float16)0.f;
    }
  int rbase = (lane & 16) ? 8 : 0;
  for (int tile = wave; tile < 20; tile += 4) {
    int r = tile * 16 + col;
    int nl = r / KNN_;
    v8f acc = {};
#pragma unroll
    for (int kc = 0; kc < KCH; kc++) {
      v16h a;
#pragma unroll
      for (int e = 0; e < 16; e++) {
        int ka = kc * 32 + wmma_a_k(lane, e);
        _Float16 av;
        if (ka < CIN)      av = nbf[r][ka];
        else if (ka < CE)  av = ctrs[nl][ka - CIN];
        else               av = (_Float16)0.f;
        a[e] = av;
      }
      acc = __builtin_amdgcn_wmma_f32_16x16x32_f16(false, a, false, bfr[kc], (short)0, acc, false, false);
    }
#pragma unroll
    for (int rr = 0; rr < 8; rr++) {
      int row = tile * 16 + rr + rbase;
      obuf[row][col] = lrelu(acc[rr] * s + sh);
    }
  }
  __syncthreads();
  for (int t = threadIdx.x; t < 256; t += blockDim.x) {
    int nl = t >> 4, oc = t & 15;
    float mx = NEG_INF_;
#pragma unroll
    for (int k = 0; k < KNN_; k++) mx = fmaxf(mx, obuf[nl * KNN_ + k][oc]);
    outmax[((size_t)b * COUT + o0 + oc) * N_ + n0 + nl] = mx;
  }
}

// ---------------- nb-branch: fold conv1x4+mean_k into K=4*CIN GEMM ----------
template <int CIN, int O>
__global__ void nbweff_kernel(const float* __restrict__ w /*(O,CIN,1,4)*/, const float* __restrict__ cb,
    const float* g, const float* bb_, const float* m, const float* v,
    float* __restrict__ weff /* O*4CIN weights, then alpha[O], beta[O] */) {
  int tid = blockIdx.x * blockDim.x + threadIdx.x;
  if (tid < O * CIN) {
    int o = tid / CIN, i = tid % CIN;
    const float* wp = w + (size_t)(o * CIN + i) * 4;
    float W0 = wp[0], W1 = wp[1], W2 = wp[2], W3 = wp[3];
    float S = W0 + W1 + W2 + W3;
    weff[o * 4 * CIN + 0 * CIN + i] = S;          // * sum_k pn[k]
    weff[o * 4 * CIN + 1 * CIN + i] = -(W2 + W3); // * pn[0]
    weff[o * 4 * CIN + 2 * CIN + i] = -W3;        // * pn[1]
    weff[o * 4 * CIN + 3 * CIN + i] = -W0;        // * pn[K-1]
  }
  if (tid < O) {
    float s = g[tid] * rsqrtf(v[tid] + EPS_);
    float sh = bb_[tid] - m[tid] * s;
    weff[O * 4 * CIN + tid]     = s / (float)KNN_;   // alpha
    weff[O * 4 * CIN + O + tid] = s * cb[tid] + sh;  // beta
  }
}

template <int CIN, int O>
__global__ void nb_apply_kernel(const float* __restrict__ src, const int* __restrict__ idx,
    const float* __restrict__ weff, float* __restrict__ out /*(B,O,N) += */) {
  __shared__ float U[16][4 * CIN];
  int b = blockIdx.y;
  int n0 = blockIdx.x * 16;
  const float* sb = src + (size_t)b * CIN * N_;
  const int* ib = idx + ((size_t)b * N_ + n0) * KNN_;
  for (int t = threadIdx.x; t < 16 * CIN; t += blockDim.x) {
    int nl = t / CIN, i = t % CIN;
    const float* row = sb + i * N_;
    const int* jj = ib + nl * KNN_;
    float sum = 0.f, p0 = 0.f, p1 = 0.f, pl = 0.f;
#pragma unroll
    for (int k = 0; k < KNN_; k++) {
      float val = row[jj[k]];
      sum += val;
      if (k == 0) p0 = val;
      if (k == 1) p1 = val;
      if (k == KNN_ - 1) pl = val;
    }
    U[nl][i] = sum; U[nl][CIN + i] = p0; U[nl][2 * CIN + i] = p1; U[nl][3 * CIN + i] = pl;
  }
  __syncthreads();
  const float* alpha = weff + O * 4 * CIN;
  const float* beta = alpha + O;
  for (int t = threadIdx.x; t < 16 * O; t += blockDim.x) {
    int nl = t / O, o = t % O;
    float acc = 0.f;
    const float* wr = weff + o * 4 * CIN;
#pragma unroll 4
    for (int c = 0; c < 4 * CIN; c++) acc += wr[c] * U[nl][c];
    // single writer per output element -> plain RMW, deterministic
    float* dst = &out[((size_t)b * O + o) * N_ + n0 + nl];
    *dst = *dst + alpha[o] * acc + beta[o];
  }
}

// ---------------- conv5 (1024x320): partial max + partial sums over N -------
__global__ void conv5_kernel(const float* __restrict__ x1, const float* __restrict__ x2,
    const float* __restrict__ x3, const float* __restrict__ x4,
    const float* __restrict__ W /*1024x320*/,
    const float* g, const float* bb_, const float* m, const float* v,
    float* __restrict__ pmax, float* __restrict__ psum /*(B,1024,N/16,2)*/) {
  int b = blockIdx.x / (N_ / 16);
  int nch = blockIdx.x % (N_ / 16);
  int n0 = nch * 16;
  int o0 = blockIdx.y * 16;
  int lane = threadIdx.x & 31;
  int col = lane & 15;
  int o = o0 + col;
  for (int t = lane * 64; t < 320 * 16 * 4; t += 32 * 64)
    __builtin_prefetch((const char*)(W + o0 * 320) + t, 0, 0);
  v8f acc = {};
#pragma unroll
  for (int kc = 0; kc < 10; kc++) {
    v16h a, bf;
#pragma unroll
    for (int e = 0; e < 16; e++) {
      int ka = kc * 32 + wmma_a_k(lane, e);
      const float* sp; int c, chs;
      if (ka < 64)       { sp = x1; c = ka;       chs = 64; }
      else if (ka < 128) { sp = x2; c = ka - 64;  chs = 64; }
      else if (ka < 192) { sp = x3; c = ka - 128; chs = 64; }
      else               { sp = x4; c = ka - 192; chs = 128; }
      a[e]  = (_Float16)sp[((size_t)b * chs + c) * N_ + n0 + col];
      bf[e] = (_Float16)W[o * 320 + kc * 32 + wmma_b_k(lane, e)];
    }
    acc = __builtin_amdgcn_wmma_f32_16x16x32_f16(false, a, false, bf, (short)0, acc, false, false);
  }
  float s = g[o] * rsqrtf(v[o] + EPS_);
  float sh = bb_[o] - m[o] * s;
  float mx = NEG_INF_, sm = 0.f;
#pragma unroll
  for (int r = 0; r < 8; r++) { float z = lrelu(acc[r] * s + sh); mx = fmaxf(mx, z); sm += z; }
  int half = (lane & 16) ? 1 : 0;
  size_t slot = (((size_t)b * 1024 + o) * (N_ / 16) + nch) * 2 + half;
  pmax[slot] = mx;
  psum[slot] = sm;
}

// ---------------- final MLP head --------------------------------------------
__global__ void final_mlp_kernel(const float* __restrict__ gmax, const float* __restrict__ gsum,
    const float* __restrict__ L1, const float* g6, const float* b6, const float* m6, const float* v6,
    const float* __restrict__ L2, const float* __restrict__ L2b,
    const float* g7, const float* b7, const float* m7, const float* v7,
    const float* __restrict__ L3, const float* __restrict__ L3b, float* __restrict__ out) {
  __shared__ float gv[2048], h1s[512], h2s[256];
  int b = blockIdx.x;
  for (int t = threadIdx.x; t < 1024; t += blockDim.x) {
    gv[t] = gmax[b * 1024 + t];
    gv[1024 + t] = gsum[b * 1024 + t] * (1.f / (float)N_);
  }
  __syncthreads();
  for (int o = threadIdx.x; o < 512; o += blockDim.x) {
    float acc = 0.f;
    for (int c = 0; c < 2048; c++) acc += L1[o * 2048 + c] * gv[c];
    float s = g6[o] * rsqrtf(v6[o] + EPS_);
    h1s[o] = lrelu(acc * s + (b6[o] - m6[o] * s));
  }
  __syncthreads();
  for (int o = threadIdx.x; o < 256; o += blockDim.x) {
    float acc = L2b[o];
    for (int c = 0; c < 512; c++) acc += L2[o * 512 + c] * h1s[c];
    float s = g7[o] * rsqrtf(v7[o] + EPS_);
    h2s[o] = lrelu(acc * s + (b7[o] - m7[o] * s));
  }
  __syncthreads();
  if (threadIdx.x < 40) {
    float acc = L3b[threadIdx.x];
    for (int c = 0; c < 256; c++) acc += L3[threadIdx.x * 256 + c] * h2s[c];
    out[b * 40 + threadIdx.x] = acc;
  }
}

// ---------------- host orchestration ----------------------------------------
extern "C" void kernel_launch(void* const* d_in, const int* in_sizes, int n_in,
                              void* d_out, int out_size, void* d_ws, size_t ws_size,
                              hipStream_t stream) {
  (void)in_sizes; (void)n_in; (void)out_size; (void)ws_size;
  auto F = [&](int i) { return (const float*)d_in[i]; };
  // input order (setup_inputs dict order; each bn dict flattens g,b,m,v):
  // 0 x | 1 t_conv1_w | 2-5 t_bn1 | 6 t_conv2_w | 7-10 t_bn2 | 11 t_conv3_w |
  // 12-15 t_bn3a | 16 t_lin1_w | 17-20 t_bn3 | 21 t_lin2_w | 22-25 t_bn4 |
  // 26 t_transform_w | 27 t_transform_b | 28 conv1_w | 29-32 bn1 | 33 conv2_w |
  // 34-37 bn2 | 38 conv3_w | 39-42 bn3 | 43 conv4_w | 44-47 bn4 | 48 conv5_w |
  // 49-52 bn5 | 53 conv10_w | 54 conv10_b | 55-58 bn10 | 59 conv11_w | 60 conv11_b |
  // 61-64 bn11 | 65 conv12_w | 66 conv12_b | 67-70 bn12 | 71 conv13_w | 72 conv13_b |
  // 73-76 bn13 | 77 lin1_w | 78-81 bn6 | 82 lin2_w | 83 lin2_b | 84-87 bn7 |
  // 88 lin3_w | 89 lin3_b
  const float* x = F(0);

  char* base = (char*)d_ws;
  size_t off = 0;
  auto alloc = [&](size_t bytes) -> void* {
    void* p = base + off;
    off = (off + bytes + 255) & ~(size_t)255;
    return p;
  };
  const size_t idxB = (size_t)B_ * N_ * KNN_ * sizeof(int);
  int* idx0 = (int*)alloc(idxB);
  int* idx1 = (int*)alloc(idxB);
  int* idx2 = (int*)alloc(idxB);
  int* idx3 = (int*)alloc(idxB);
  int* idx4 = (int*)alloc(idxB);
  float* xrot  = (float*)alloc((size_t)B_ * 3 * N_ * 4);
  float* tmax  = (float*)alloc((size_t)B_ * 128 * N_ * 4);
  float* tglob = (float*)alloc((size_t)B_ * 1024 * 4);
  float* tmat  = (float*)alloc((size_t)B_ * 9 * 4);
  float* x1b = (float*)alloc((size_t)B_ * 64 * N_ * 4);
  float* x2b = (float*)alloc((size_t)B_ * 64 * N_ * 4);
  float* x3b = (float*)alloc((size_t)B_ * 64 * N_ * 4);
  float* x4b = (float*)alloc((size_t)B_ * 128 * N_ * 4);
  float* weff10 = (float*)alloc((64 * 12 + 2 * 64) * 4);
  float* weff11 = (float*)alloc((64 * 12 + 2 * 64) * 4);
  float* weff12 = (float*)alloc((64 * 256 + 2 * 64) * 4);
  float* weff13 = (float*)alloc((128 * 256 + 2 * 128) * 4);
  float* gmax = (float*)alloc((size_t)B_ * 1024 * 4);
  float* gsum = (float*)alloc((size_t)B_ * 1024 * 4);
  float* psum = (float*)alloc((size_t)B_ * 1024 * (N_ / 16) * 2 * 4);
  float* pmax = (float*)alloc((size_t)B_ * 1024 * (N_ / 16) * 2 * 4);
  float* tpmax = (float*)alloc((size_t)B_ * 1024 * (N_ / 16) * 2 * 4);

  dim3 knnGrid(N_ / 256, B_);
  dim3 tileGrid(B_ * N_ / 16, 8);

  // 1) kNN on raw x -> idx0; tnet
  knn_kernel<3, 512><<<knnGrid, 256, 0, stream>>>(x, idx0);
  tnet_edge_kernel<<<tileGrid, 128, 0, stream>>>(x, idx0,
      F(1), F(2), F(3), F(4), F(5),
      F(6), F(7), F(8), F(9), F(10), tmax);
  tnet_conv3_kernel<<<dim3(B_ * N_ / 16, 64), 32, 0, stream>>>(tmax,
      F(11), F(12), F(13), F(14), F(15), tpmax);
  reduce_max_kernel<<<dim3((B_ * 1024 + 255) / 256), 256, 0, stream>>>(tpmax, tglob);
  tnet_mlp_kernel<<<dim3(B_), 256, 0, stream>>>(tglob,
      F(16), F(17), F(18), F(19), F(20),
      F(21), F(22), F(23), F(24), F(25),
      F(26), F(27), tmat);
  rotate_kernel<<<knnGrid, 256, 0, stream>>>(x, tmat, xrot);

  // 2) stage 1: conv1 on edges of rotated x + nb(pn0 = raw x @ idx0, conv10)
  knn_kernel<3, 512><<<knnGrid, 256, 0, stream>>>(xrot, idx1);
  edgeconv_kernel<3, 64><<<dim3(B_ * N_ / 16, 4), 128, 0, stream>>>(xrot, idx1,
      F(28), F(29), F(30), F(31), F(32), x1b);
  nbweff_kernel<3, 64><<<dim3(1), 256, 0, stream>>>(F(53), F(54), F(55), F(56), F(57), F(58), weff10);
  nb_apply_kernel<3, 64><<<dim3(N_ / 16, B_), 256, 0, stream>>>(x, idx0, weff10, x1b);

  // 3) stage 2
  knn_kernel<64, 128><<<knnGrid, 256, 0, stream>>>(x1b, idx2);
  edgeconv_kernel<64, 64><<<dim3(B_ * N_ / 16, 4), 128, 0, stream>>>(x1b, idx2,
      F(33), F(34), F(35), F(36), F(37), x2b);
  nbweff_kernel<3, 64><<<dim3(1), 256, 0, stream>>>(F(59), F(60), F(61), F(62), F(63), F(64), weff11);
  nb_apply_kernel<3, 64><<<dim3(N_ / 16, B_), 256, 0, stream>>>(xrot, idx1, weff11, x2b);

  // 4) stage 3
  knn_kernel<64, 128><<<knnGrid, 256, 0, stream>>>(x2b, idx3);
  edgeconv_kernel<64, 64><<<dim3(B_ * N_ / 16, 4), 128, 0, stream>>>(x2b, idx3,
      F(38), F(39), F(40), F(41), F(42), x3b);
  nbweff_kernel<64, 64><<<dim3((64 * 64 + 255) / 256), 256, 0, stream>>>(
      F(65), F(66), F(67), F(68), F(69), F(70), weff12);
  nb_apply_kernel<64, 64><<<dim3(N_ / 16, B_), 256, 0, stream>>>(x1b, idx2, weff12, x3b);

  // 5) stage 4
  knn_kernel<64, 128><<<knnGrid, 256, 0, stream>>>(x3b, idx4);
  edgeconv_kernel<64, 128><<<dim3(B_ * N_ / 16, 8), 128, 0, stream>>>(x3b, idx4,
      F(43), F(44), F(45), F(46), F(47), x4b);
  nbweff_kernel<64, 128><<<dim3((128 * 64 + 255) / 256), 256, 0, stream>>>(
      F(71), F(72), F(73), F(74), F(75), F(76), weff13);
  nb_apply_kernel<64, 128><<<dim3(N_ / 16, B_), 256, 0, stream>>>(x2b, idx3, weff13, x4b);

  // 6) conv5 + global max / ordered mean
  conv5_kernel<<<dim3(B_ * N_ / 16, 64), 32, 0, stream>>>(x1b, x2b, x3b, x4b,
      F(48), F(49), F(50), F(51), F(52), pmax, psum);
  reduce_maxsum_kernel<<<dim3((B_ * 1024 + 255) / 256), 256, 0, stream>>>(pmax, psum, gmax, gsum);

  // 7) head
  final_mlp_kernel<<<dim3(B_), 256, 0, stream>>>(gmax, gsum,
      F(77), F(78), F(79), F(80), F(81),
      F(82), F(83), F(84), F(85), F(86), F(87),
      F(88), F(89), (float*)d_out);
}
